// WeightedGNN_91164975825445
// MI455X (gfx1250) — matrix operand order repaired and checked by reading, not compiled, not run on hardware
//
#include <hip/hip_runtime.h>
#include <hip/hip_bf16.h>

// ---------------------------------------------------------------------------
// WeightedGNN for MI455X (gfx1250): f16 WMMA GEMMs + fp32 epilogues.
// All WMMA B operands pre-packed into fragment-native layout (1 v16h/lane/tile)
// so inner loops are pure global_load_b128 + v_wmma.
// ---------------------------------------------------------------------------

typedef __attribute__((ext_vector_type(16))) _Float16 v16h;
typedef __attribute__((ext_vector_type(8)))  _Float16 v8h;
typedef __attribute__((ext_vector_type(8)))  float    v8f;

#define NB 16      // batch
#define NL 64      // seq len
#define ND 256     // hidden
#define NR 8       // relation types
#define BL (NB*NL) // 1024 rows

// ---- WMMA fragment helpers (wave32; A/B in contiguous k order, C/D as in CDNA5 ISA) ----

// A-matrix 16x32 f16 from row-major memory, row stride `stride` halves.
// lane<16: M=lane, K=0..15 ; lane>=16: M=lane-16, K=16..31
__device__ __forceinline__ v16h load_frag_a(const _Float16* base, int stride) {
  int lane = threadIdx.x & 31;
  int row  = lane & 15;
  int kb   = (lane < 16) ? 0 : 16;
  const _Float16* p = base + (size_t)row * stride + kb;
  union { v16h v; struct { v8h lo, hi; } s; } u;
  u.s.lo = *(const v8h*)(p);
  u.s.hi = *(const v8h*)(p + 8);
  return u.v;
}

// B-matrix 32x16 f16 from packed layout: tile = 32 lanes x 16 contiguous halves.
__device__ __forceinline__ v16h load_frag_b_packed(const _Float16* packed, int tileIdx) {
  int lane = threadIdx.x & 31;
  return *((const v16h*)packed + (size_t)tileIdx * 32 + lane);
}

__device__ __forceinline__ v8f wmma16(v16h a, v16h b, v8f c) {
  return __builtin_amdgcn_wmma_f32_16x16x32_f16(false, a, false, b, (short)0, c,
                                                false, false);
}

// C/D fragment: lane<16 -> (M=r, N=lane); lane>=16 -> (M=8+r, N=lane-16)

// ---------------------------- setup kernels --------------------------------

__global__ void k_lengths(const int* __restrict__ tokens, int* __restrict__ lengths) {
  int b = threadIdx.x;
  if (b < NB) {
    int c = 0;
    for (int i = 0; i < NL; ++i) c += (tokens[i * NB + b] != 0) ? 1 : 0;
    lengths[b] = c;
  }
}

// Xh[r, d] = f16(emb[tok(b,i), d]),  r = b*64 + i, tokens layout [L,B]
__global__ void k_gather(const int* __restrict__ tokens, const float* __restrict__ emb,
                         _Float16* __restrict__ Xh) {
  int r = blockIdx.x, d = threadIdx.x;
  int b = r >> 6, i = r & 63;
  int tok = tokens[i * NB + b];
  Xh[(size_t)r * ND + d] = (_Float16)emb[(size_t)tok * ND + d];
}

// Pack a row-major f32 W[K][N] into fragment-native f16 B layout.
// element (k,n): kt=k>>5, kq=k&31, nt=n>>4, nc=n&15
// out[ ((nt*KT + kt)*32 + lane)*16 + ele ],  lane = nc + 16*(kq>=16), ele = kq&15
__global__ void k_pack_b(const float* __restrict__ W, _Float16* __restrict__ out,
                         int K, int N) {
  int idx = blockIdx.x * 256 + threadIdx.x;
  if (idx >= K * N) return;
  int k = idx / N, n = idx - k * N;
  int kt = k >> 5, kq = k & 31, nt = n >> 4, nc = n & 15;
  int lane = nc + ((kq & 16) ? 16 : 0);
  int ele  = kq & 15;
  int KT   = K >> 5;
  out[((((size_t)nt * KT + kt) * 32 + lane) << 4) + ele] = (_Float16)W[idx];
}

// At[b][j][i*8+h] = A_rels[b][i][j][h]  (f32 -> f16): GEMM A operand, K contiguous
__global__ void k_transposeA(const float* __restrict__ A, _Float16* __restrict__ At) {
  int n = blockIdx.x * 256 + threadIdx.x;       // enumerates [b][j][i][h]
  int h = n & 7, i = (n >> 3) & 63, j = (n >> 9) & 63, b = n >> 15;
  At[n] = (_Float16)A[(((size_t)(b * 64 + i) * 64 + j) * 8) + h];
}

// in3[r, 0:768] = X[r] @ W_in + b_in   (hoisted; shift|scale|proj_x)
__global__ void k_inproj(const _Float16* __restrict__ Xh, const _Float16* __restrict__ Winp,
                         const float* __restrict__ b_in, float* __restrict__ in3) {
  int rt = blockIdx.x;                       // 64 row tiles of 16
  int w = threadIdx.x >> 5, lane = threadIdx.x & 31;
  int col = lane & 15, rowb = (lane < 16) ? 0 : 8;
  v8f acc[6] = {};
#pragma unroll
  for (int kt = 0; kt < 8; ++kt) {           // K = 256
    v16h a = load_frag_a(Xh + (size_t)(rt * 16) * ND + kt * 32, ND);
#pragma unroll
    for (int t = 0; t < 6; ++t) {            // 48 n-tiles / 8 waves
      int nt = w * 6 + t;
      acc[t] = wmma16(a, load_frag_b_packed(Winp, nt * 8 + kt), acc[t]);
    }
  }
#pragma unroll
  for (int t = 0; t < 6; ++t) {
    int c = (w * 6 + t) * 16 + col;
    float bias = b_in[c];
#pragma unroll
    for (int r = 0; r < 8; ++r) {
      int row = rt * 16 + rowb + r;
      in3[(size_t)row * 768 + c] = acc[t][r] + bias;
    }
  }
}

// ---------------------------- per-step kernels -----------------------------

// P[b,i,h,:] = LN_256( (prev @ W_proj + b_proj)[., h*256:(h+1)*256] )
// Output written directly in fragment-packed B layout for k_agg (K = i*8+h).
// grid (64 row-tiles, 4 column-quarters of 512); each quarter = 2 whole rels.
__global__ void k_proj_ln(const _Float16* __restrict__ prevh, const _Float16* __restrict__ Wpp,
                          const float* __restrict__ bproj, _Float16* __restrict__ Php) {
  __shared__ float tc[16][512];
  __shared__ float ps[32][8], ps2[32][8];
  __shared__ float mn[32], rs[32];
  int rt = blockIdx.x, cq = blockIdx.y;
  int w = threadIdx.x >> 5, lane = threadIdx.x & 31;
  int col = lane & 15, rowb = (lane < 16) ? 0 : 8;
  v8f acc[4] = {};
#pragma unroll
  for (int kt = 0; kt < 8; ++kt) {           // K = 256
    v16h a = load_frag_a(prevh + (size_t)(rt * 16) * ND + kt * 32, ND);
#pragma unroll
    for (int t = 0; t < 4; ++t) {            // 32 local n-tiles / 8 waves
      int nt = cq * 32 + w * 4 + t;          // global n-tile (N = 2048, KT = 8)
      acc[t] = wmma16(a, load_frag_b_packed(Wpp, nt * 8 + kt), acc[t]);
    }
  }
#pragma unroll
  for (int t = 0; t < 4; ++t) {
    int lt = w * 4 + t;
    float bias = bproj[cq * 512 + lt * 16 + col];
#pragma unroll
    for (int r = 0; r < 8; ++r) tc[rowb + r][lt * 16 + col] = acc[t][r] + bias;
  }
  __syncthreads();
  // LayerNorm: 32 groups (16 rows x 2 local rels) of 256 elements
  int g = threadIdx.x >> 3, sub = threadIdx.x & 7;
  int row = g >> 1, rl = g & 1;
  float s = 0.f, s2 = 0.f;
  for (int t = 0; t < 32; ++t) {
    float x = tc[row][rl * 256 + sub * 32 + t];
    s += x; s2 += x * x;
  }
  ps[g][sub] = s; ps2[g][sub] = s2;
  __syncthreads();
  if (sub == 0) {
    float a = 0.f, b2 = 0.f;
    for (int t = 0; t < 8; ++t) { a += ps[g][t]; b2 += ps2[g][t]; }
    float m = a * (1.f / 256.f);
    float v = b2 * (1.f / 256.f) - m * m;
    mn[g] = m; rs[g] = rsqrtf(v + 1e-5f);
  }
  __syncthreads();
  float m = mn[g], r = rs[g];
  int grow = rt * 16 + row;                  // == b*64 + i
  int b = grow >> 6, i = grow & 63;
  int hrel = cq * 2 + rl;
  int kk = i * NR + hrel;                    // K row of the agg GEMM (0..511)
  int kt = kk >> 5, kq = kk & 31;
  int lane16 = (kq & 16) ? 16 : 0;
  int ele = kq & 15;
  _Float16* dstb = Php + (size_t)b * (512 * ND);
  for (int t = 0; t < 32; ++t) {
    int d = sub * 32 + t;
    int nt = d >> 4, nc = d & 15;            // N = 256, KT = 16
    size_t off = ((((size_t)nt * 16 + kt) * 32) + (nc + lane16)) * 16 + ele;
    dstb[off] = (_Float16)((tc[row][rl * 256 + d] - m) * r);
  }
}

// lin[b,j,:] = At[b] (16j x 512k) @ P[b] (512k x 256d); then shrink + LN + gate
// grid 64: b = blk>>2, jt = blk&3
__global__ void k_agg(const _Float16* __restrict__ Ath, const _Float16* __restrict__ Php,
                      const float* __restrict__ in3, _Float16* __restrict__ hbuf) {
  __shared__ float tc[16][256];
  __shared__ float ps[16][16], ps2[16][16];
  __shared__ float mn[16], rs[16];
  int b = blockIdx.x >> 2, jt = blockIdx.x & 3;
  int w = threadIdx.x >> 5, lane = threadIdx.x & 31;
  int col = lane & 15, rowb = (lane < 16) ? 0 : 8;
  const _Float16* Ab  = Ath + ((size_t)b * NL + jt * 16) * 512;
  const _Float16* Phb = Php + (size_t)b * (512 * ND);
  v8f acc[2] = {};
#pragma unroll
  for (int kt = 0; kt < 16; ++kt) {          // K = 512
    v16h a = load_frag_a(Ab + kt * 32, 512);
#pragma unroll
    for (int t = 0; t < 2; ++t) {            // 16 n-tiles / 8 waves
      int nt = w * 2 + t;
      acc[t] = wmma16(a, load_frag_b_packed(Phb, nt * 16 + kt), acc[t]);
    }
  }
#pragma unroll
  for (int t = 0; t < 2; ++t) {
    int n0 = (w * 2 + t) * 16;
#pragma unroll
    for (int r = 0; r < 8; ++r) tc[rowb + r][n0 + col] = acc[t][r];
  }
  __syncthreads();
  // t = proj_x + shrink(lin); LN over 256; h = relu(shift*ln + scale)
  int g = threadIdx.x >> 4, sub = threadIdx.x & 15;
  int grow = b * NL + jt * 16 + g;
  const float* rowp = in3 + (size_t)grow * 768;  // [shift|scale|proj_x]
  float s = 0.f, s2 = 0.f;
  for (int t = 0; t < 16; ++t) {
    int d = sub * 16 + t;
    float lin = tc[g][d];
    float tt = rowp[512 + d] + (lin - tanhf(lin));
    tc[g][d] = tt;
    s += tt; s2 += tt * tt;
  }
  ps[g][sub] = s; ps2[g][sub] = s2;
  __syncthreads();
  if (sub == 0) {
    float a = 0.f, b2 = 0.f;
    for (int t = 0; t < 16; ++t) { a += ps[g][t]; b2 += ps2[g][t]; }
    float m = a * (1.f / 256.f);
    float v = b2 * (1.f / 256.f) - m * m;
    mn[g] = m; rs[g] = rsqrtf(v + 1e-5f);
  }
  __syncthreads();
  float m = mn[g], r = rs[g];
  for (int t = 0; t < 16; ++t) {
    int d = sub * 16 + t;
    float ln = (tc[g][d] - m) * r;
    float hv = rowp[d] * ln + rowp[256 + d];
    hbuf[(size_t)grow * ND + d] = (_Float16)fmaxf(hv, 0.f);
  }
}

// y = tanh(h @ W_out + b_out), zeroed per-batch where step_i > lengths[b]
__global__ void k_out(const _Float16* __restrict__ hbuf, const _Float16* __restrict__ Wop,
                      const float* __restrict__ bout, const int* __restrict__ lengths,
                      int step_i, float* __restrict__ prevf, _Float16* __restrict__ prevh) {
  int rt = blockIdx.x;                       // 64 row tiles
  int w = threadIdx.x >> 5, lane = threadIdx.x & 31;
  int col = lane & 15, rowb = (lane < 16) ? 0 : 8;
  v8f acc[2] = {};
#pragma unroll
  for (int kt = 0; kt < 8; ++kt) {           // K = 256
    v16h a = load_frag_a(hbuf + (size_t)(rt * 16) * ND + kt * 32, ND);
#pragma unroll
    for (int t = 0; t < 2; ++t) {            // 16 n-tiles / 8 waves
      int nt = w * 2 + t;
      acc[t] = wmma16(a, load_frag_b_packed(Wop, nt * 8 + kt), acc[t]);
    }
  }
#pragma unroll
  for (int t = 0; t < 2; ++t) {
    int n0 = (w * 2 + t) * 16;
    float bias = bout[n0 + col];
#pragma unroll
    for (int r = 0; r < 8; ++r) {
      int row = rt * 16 + rowb + r;
      int b = row >> 6;
      float y = (step_i > lengths[b]) ? 0.f : tanhf(acc[t][r] + bias);
      size_t o = (size_t)row * ND + n0 + col;
      prevf[o] = y;
      prevh[o] = (_Float16)y;
    }
  }
}

// final[b,d] = sum_i prev[b,i,d] * root[b,i]
__global__ void k_final(const float* __restrict__ prevf, const float* __restrict__ root,
                        float* __restrict__ out) {
  int b = blockIdx.x, d = threadIdx.x;
  float acc = 0.f;
  for (int i = 0; i < NL; ++i)
    acc += prevf[((size_t)(b * NL + i)) * ND + d] * root[b * NL + i];
  out[b * ND + d] = acc;
}

// ------------------------------- launcher ----------------------------------

extern "C" void kernel_launch(void* const* d_in, const int* in_sizes, int n_in,
                              void* d_out, int out_size, void* d_ws, size_t ws_size,
                              hipStream_t stream) {
  (void)in_sizes; (void)n_in; (void)out_size; (void)ws_size;
  const int*   tokens = (const int*)d_in[0];
  const float* A_rels = (const float*)d_in[1];
  const float* root   = (const float*)d_in[2];
  const float* emb    = (const float*)d_in[3];
  const float* Wproj  = (const float*)d_in[4];
  const float* bproj  = (const float*)d_in[5];
  const float* Win    = (const float*)d_in[6];
  const float* b_in   = (const float*)d_in[7];
  const float* Wout   = (const float*)d_in[8];
  const float* bout   = (const float*)d_in[9];

  char* w = (char*)d_ws;
  auto alloc = [&](size_t bytes) -> void* {
    void* p = (void*)w;
    w += (bytes + 255) & ~(size_t)255;
    return p;
  };
  int*      lengths = (int*)alloc(NB * sizeof(int));
  _Float16* Xh    = (_Float16*)alloc((size_t)BL * ND * 2);
  _Float16* Winp  = (_Float16*)alloc((size_t)ND * 768 * 2);   // packed B
  _Float16* Wpp   = (_Float16*)alloc((size_t)ND * 2048 * 2);  // packed B
  _Float16* Wop   = (_Float16*)alloc((size_t)ND * ND * 2);    // packed B
  _Float16* Ath   = (_Float16*)alloc((size_t)NB * NL * 512 * 2);
  float*    in3   = (float*)alloc((size_t)BL * 768 * 4);
  _Float16* Php   = (_Float16*)alloc((size_t)NB * 512 * ND * 2); // packed B
  _Float16* hb    = (_Float16*)alloc((size_t)BL * ND * 2);
  float*    prevf = (float*)alloc((size_t)BL * ND * 4);
  _Float16* prevh = (_Float16*)alloc((size_t)BL * ND * 2);

  // h_prev initial state = 0 (graph-capturable memset nodes)
  hipMemsetAsync(prevh, 0, (size_t)BL * ND * 2, stream);
  hipMemsetAsync(prevf, 0, (size_t)BL * ND * 4, stream);

  // one-time setup
  k_lengths<<<1, 32, 0, stream>>>(tokens, lengths);
  k_gather<<<BL, ND, 0, stream>>>(tokens, emb, Xh);
  k_pack_b<<<(ND * 768 + 255) / 256, 256, 0, stream>>>(Win, Winp, ND, 768);
  k_pack_b<<<(ND * 2048 + 255) / 256, 256, 0, stream>>>(Wproj, Wpp, ND, 2048);
  k_pack_b<<<(ND * ND + 255) / 256, 256, 0, stream>>>(Wout, Wop, ND, ND);
  k_transposeA<<<(NB * NL * NL * NR) / 256, 256, 0, stream>>>(A_rels, Ath);
  k_inproj<<<64, 256, 0, stream>>>(Xh, Winp, b_in, in3);

  // serial scan: i = 64 .. 1
  for (int s = 0; s < NL; ++s) {
    int step_i = NL - s;
    k_proj_ln<<<dim3(64, 4), 256, 0, stream>>>(prevh, Wpp, bproj, Php);
    k_agg<<<64, 256, 0, stream>>>(Ath, Php, in3, hb);
    k_out<<<64, 256, 0, stream>>>(hb, Wop, bout, lengths, step_i, prevf, prevh);
  }

  k_final<<<NB, ND, 0, stream>>>(prevf, root, (float*)d_out);
}
